// PKaGNN_59511066853668
// MI455X (gfx1250) — matrix operand order repaired and checked by reading, not compiled, not run on hardware
//
#include <hip/hip_runtime.h>
#include <hip/hip_fp16.h>

// Model dims
#define HD   128
#define ADIM 25
#define EDIM 10
#define RDIM 64
#define NLAY 4

typedef __attribute__((ext_vector_type(16))) _Float16 v16h;
typedef __attribute__((ext_vector_type(8)))  _Float16 v8h;
typedef __attribute__((ext_vector_type(8)))  float    v8f;

__device__ __forceinline__ float gelu_f(float x){
  float t = 0.7978845608028654f * (x + 0.044715f * x * x * x);
  return 0.5f * x * (1.0f + tanhf(t));
}

__device__ __forceinline__ v16h cat8(v8h lo, v8h hi){
  return __builtin_shufflevector(lo, hi, 0,1,2,3,4,5,6,7,8,9,10,11,12,13,14,15);
}

// A fragment (16x32 f16, M x K): half j of lane L holds
// k = kbase + (j>>3)*16 + (L>>4)*8 + (j&7), m = L&15  -> two contiguous 16B loads
__device__ __forceinline__ v16h frag_a(const _Float16* sA, int lane, int kb, int ldk){
  int m = lane & 15, g = lane >> 4;
  const _Float16* p = sA + m * ldk + kb + g * 8;
  return cat8(*(const v8h*)p, *(const v8h*)(p + 16));
}

// B fragment (32x16 f16, K x N) from column-major (transposed [N][K]) weights:
// half j of lane L holds k = kbase + (j>>3)*16 + (L>>4)*8 + (j&7), n = L&15 -> two contiguous 16B loads
__device__ __forceinline__ v16h frag_b(const _Float16* Bt, int lane, int kb, int K){
  int n = lane & 15, g = lane >> 4;
  const _Float16* p = Bt + n * K + kb + g * 8;
  return cat8(*(const v8h*)p, *(const v8h*)(p + 16));
}

// 16x16 output tile = A[16xK] (LDS, row-major f16) x B[KxN-tile] (global f16 transposed)
template<int K, int LDK>
__device__ __forceinline__ v8f gemm16(const _Float16* sA, const _Float16* BtN, int lane){
  v8f acc = {0.f,0.f,0.f,0.f,0.f,0.f,0.f,0.f};
  #pragma unroll
  for (int kb = 0; kb < K; kb += 32){
    v16h a = frag_a(sA, lane, kb, LDK);
    v16h b = frag_b(BtN, lane, kb, K);
    acc = __builtin_amdgcn_wmma_f32_16x16x32_f16(false, a, false, b, (short)0, acc,
                                                 false, false);
  }
  return acc;
}

// ---------------- utility kernels ----------------

__global__ void fill0_kernel(float* p, int n){
  int i = blockIdx.x * blockDim.x + threadIdx.x;
  if (i < n) p[i] = 0.0f;
}

// src: f32 row-major [K][Nc]  ->  dst: f16 [Nc][K]
__global__ void to_f16T_kernel(const float* __restrict__ src, _Float16* __restrict__ dst,
                               int K, int Nc){
  int i = blockIdx.x * blockDim.x + threadIdx.x;
  if (i < K * Nc){
    int k = i / Nc, n = i % Nc;
    dst[n * K + k] = (_Float16)src[i];
  }
}

__global__ void count_kernel(const int* __restrict__ ei, float* __restrict__ cnt, int En){
  int e = blockIdx.x * blockDim.x + threadIdx.x;
  if (e < En) atomicAdd(&cnt[ei[En + e]], 1.0f);
}

// ---------------- encoder: h = gelu(x@w1+b1)@w2 + b2 ----------------

__global__ void encoder_kernel(const float* __restrict__ x,
                               const float* __restrict__ w1, const float* __restrict__ b1,
                               const _Float16* __restrict__ w2T, const float* __restrict__ b2,
                               float* __restrict__ h, _Float16* __restrict__ h16, int Nn){
  __shared__ __align__(16) _Float16 sA[16 * HD];
  __shared__ float sX[16 * ADIM];
  int tid = threadIdx.x, lane = tid & 31, wave = tid >> 5;
  int r0 = blockIdx.x * 16;

  for (int i = tid; i < 16 * ADIM; i += 256){
    int m = i / ADIM, k = i % ADIM;
    int row = r0 + m; if (row >= Nn) row = Nn - 1;
    sX[i] = x[row * ADIM + k];
  }
  __syncthreads();
  for (int i = tid; i < 16 * HD; i += 256){
    int m = i >> 7, c = i & 127;
    float a = b1[c];
    #pragma unroll
    for (int k = 0; k < ADIM; ++k) a += sX[m * ADIM + k] * w1[k * HD + c];
    sA[i] = (_Float16)gelu_f(a);
  }
  __syncthreads();

  v8f acc = gemm16<HD, HD>(sA, w2T + wave * 16 * HD, lane);
  int n = lane & 15, g = lane >> 4, col = wave * 16 + n;
  #pragma unroll
  for (int r = 0; r < 8; ++r){
    int m = r + g * 8, row = r0 + m;
    if (row < Nn){
      float v = acc[r] + b2[col];
      h[row * HD + col]   = v;
      h16[row * HD + col] = (_Float16)v;
    }
  }
}

// ------- fused per-layer edge kernel: ew, msg, scatter-add into agg -------

__global__ void edge_layer_kernel(const int* __restrict__ ei, const float* __restrict__ ea,
                                  const float* __restrict__ ew1, const float* __restrict__ eb1,
                                  const _Float16* __restrict__ ew2T, const float* __restrict__ eb2,
                                  const _Float16* __restrict__ mw1T, const float* __restrict__ mb1,
                                  const _Float16* __restrict__ mw2T, const float* __restrict__ mb2,
                                  const _Float16* __restrict__ h16, float* __restrict__ agg,
                                  int En){
  __shared__ __align__(16) _Float16 sA[16 * HD];
  __shared__ float sEA[16 * EDIM];
  int tid = threadIdx.x, lane = tid & 31, wave = tid >> 5;
  int r0 = blockIdx.x * 16;
  int n = lane & 15, g = lane >> 4, col = wave * 16 + n;

  // edge_attr tile
  for (int i = tid; i < 16 * EDIM; i += 256){
    int m = i / EDIM, k = i % EDIM;
    int row = r0 + m; if (row >= En) row = En - 1;
    sEA[i] = ea[row * EDIM + k];
  }
  __syncthreads();
  // stage1 (K=10, scalar fp32) -> gelu -> f16 A tile
  for (int i = tid; i < 16 * HD; i += 256){
    int m = i >> 7, c = i & 127;
    float a = eb1[c];
    #pragma unroll
    for (int k = 0; k < EDIM; ++k) a += sEA[m * EDIM + k] * ew1[k * HD + c];
    sA[i] = (_Float16)gelu_f(a);
  }
  __syncthreads();

  // ew = t0 @ ew2 + eb2 (kept in this wave's registers; tile matches msg tile)
  v8f ew = gemm16<HD, HD>(sA, ew2T + wave * 16 * HD, lane);
  #pragma unroll
  for (int r = 0; r < 8; ++r) ew[r] += eb2[col];
  __syncthreads();

  // gather h16[src] into A tile (one 16B vector per thread)
  {
    int m = tid >> 4;            // 0..15
    int c0 = (tid & 15) * 8;     // 0..120
    int row = r0 + m; if (row >= En) row = En - 1;
    int s = ei[row];
    *(v8h*)&sA[m * HD + c0] = *(const v8h*)&h16[s * HD + c0];
  }
  __syncthreads();

  // t1 = h_src @ mw1
  v8f t1 = gemm16<HD, HD>(sA, mw1T + wave * 16 * HD, lane);
  __syncthreads();
  #pragma unroll
  for (int r = 0; r < 8; ++r){
    int m = r + g * 8;
    sA[m * HD + col] = (_Float16)gelu_f(t1[r] + mb1[col]);
  }
  __syncthreads();

  // msg = (gelu(t1) @ mw2 + mb2) * ew ; scatter mean-sum into agg
  v8f msg = gemm16<HD, HD>(sA, mw2T + wave * 16 * HD, lane);
  #pragma unroll
  for (int r = 0; r < 8; ++r){
    int m = r + g * 8, row = r0 + m;
    if (row < En){
      float v = (msg[r] + mb2[col]) * ew[r];
      int d = ei[En + row];
      atomicAdd(&agg[d * HD + col], v);
    }
  }
}

// ------- per-layer node update: gate, residual, layernorm -------

__global__ void node_update_kernel(float* __restrict__ h, _Float16* __restrict__ h16,
                                   const float* __restrict__ agg, const float* __restrict__ cnt,
                                   const _Float16* __restrict__ gwT, const float* __restrict__ gb,
                                   const float* __restrict__ lg, const float* __restrict__ lb,
                                   int Nn){
  __shared__ __align__(16) _Float16 sA[16 * 2 * HD];
  __shared__ float sH[16 * HD];
  __shared__ float sG[16 * HD];
  __shared__ float sMu[16], sRs[16];
  int tid = threadIdx.x, lane = tid & 31, wave = tid >> 5;
  int r0 = blockIdx.x * 16;

  for (int i = tid; i < 16 * HD; i += 256){
    int m = i >> 7, k = i & 127;
    int row = r0 + m; if (row >= Nn) row = Nn - 1;
    float hv = h[row * HD + k];
    float c  = cnt[row]; c = (c < 1.0f) ? 1.0f : c;
    float gv = agg[row * HD + k] / c;
    sH[i] = hv; sG[i] = gv;
    sA[m * 2 * HD + k]      = (_Float16)hv;
    sA[m * 2 * HD + HD + k] = (_Float16)gv;
  }
  __syncthreads();

  v8f ga = gemm16<2 * HD, 2 * HD>(sA, gwT + wave * 16 * 2 * HD, lane);
  int n = lane & 15, g = lane >> 4, col = wave * 16 + n;
  #pragma unroll
  for (int r = 0; r < 8; ++r){
    int m = r + g * 8;
    float gate = 1.0f / (1.0f + expf(-(ga[r] + gb[col])));
    // this wave exclusively owns these columns of sH/sG in this phase
    sH[m * HD + col] = sH[m * HD + col] + gate * sG[m * HD + col];
  }
  __syncthreads();

  if (tid < 16){
    float mu = 0.0f;
    for (int k = 0; k < HD; ++k) mu += sH[tid * HD + k];
    mu *= (1.0f / HD);
    float var = 0.0f;
    for (int k = 0; k < HD; ++k){ float d = sH[tid * HD + k] - mu; var += d * d; }
    var *= (1.0f / HD);
    sMu[tid] = mu;
    sRs[tid] = rsqrtf(var + 1e-5f);
  }
  __syncthreads();

  for (int i = tid; i < 16 * HD; i += 256){
    int m = i >> 7, k = i & 127;
    int row = r0 + m;
    if (row < Nn){
      float v = (sH[i] - sMu[m]) * sRs[m] * lg[k] + lb[k];
      h[row * HD + k]   = v;
      h16[row * HD + k] = (_Float16)v;
    }
  }
}

// ------- output heads: out = gelu(h@w1+b1)@w2 + b2 for ion/pka/acid -------

__global__ void heads_kernel(const float* __restrict__ h,
                             const float* iw1, const float* ib1, const float* iw2, const float* ib2,
                             const float* pw1, const float* pb1, const float* pw2, const float* pb2,
                             const float* aw1, const float* ab1, const float* aw2, const float* ab2,
                             float* __restrict__ out, int Nn){
  __shared__ float sH[64 * HD];
  int tid = threadIdx.x;
  int r0 = blockIdx.x * 64;
  for (int i = tid; i < 64 * HD; i += 256){
    int m = i >> 7, k = i & 127;
    int row = r0 + m; if (row >= Nn) row = Nn - 1;
    sH[i] = h[row * HD + k];
  }
  __syncthreads();
  if (tid < 192){
    int m = tid & 63, head = tid >> 6;
    int row = r0 + m;
    const float* w1 = (head == 0) ? iw1 : ((head == 1) ? pw1 : aw1);
    const float* b1 = (head == 0) ? ib1 : ((head == 1) ? pb1 : ab1);
    const float* w2 = (head == 0) ? iw2 : ((head == 1) ? pw2 : aw2);
    const float* b2 = (head == 0) ? ib2 : ((head == 1) ? pb2 : ab2);
    float o = b2[0];
    for (int r = 0; r < RDIM; ++r){
      float a = b1[r];
      for (int k = 0; k < HD; ++k) a += sH[m * HD + k] * w1[k * RDIM + r];
      o += gelu_f(a) * w2[r];
    }
    if (row < Nn) out[head * Nn + row] = o;
  }
}

// ---------------- host side ----------------

extern "C" void kernel_launch(void* const* d_in, const int* in_sizes, int n_in,
                              void* d_out, int out_size, void* d_ws, size_t ws_size,
                              hipStream_t stream){
  const float* x         = (const float*)d_in[0];
  const int*   ei        = (const int*)  d_in[1];
  const float* edge_attr = (const float*)d_in[2];
  const float* enc_w1    = (const float*)d_in[3];
  const float* enc_b1    = (const float*)d_in[4];
  const float* enc_w2    = (const float*)d_in[5];
  const float* enc_b2    = (const float*)d_in[6];
  const float* edge_w1   = (const float*)d_in[7];
  const float* edge_b1   = (const float*)d_in[8];
  const float* edge_w2   = (const float*)d_in[9];
  const float* edge_b2   = (const float*)d_in[10];
  const float* msg_w1    = (const float*)d_in[11];
  const float* msg_b1    = (const float*)d_in[12];
  const float* msg_w2    = (const float*)d_in[13];
  const float* msg_b2    = (const float*)d_in[14];
  const float* gate_w    = (const float*)d_in[15];
  const float* gate_b    = (const float*)d_in[16];
  const float* ln_g      = (const float*)d_in[17];
  const float* ln_b      = (const float*)d_in[18];
  const float* ion_w1    = (const float*)d_in[19];
  const float* ion_b1    = (const float*)d_in[20];
  const float* ion_w2    = (const float*)d_in[21];
  const float* ion_b2    = (const float*)d_in[22];
  const float* pka_w1    = (const float*)d_in[23];
  const float* pka_b1    = (const float*)d_in[24];
  const float* pka_w2    = (const float*)d_in[25];
  const float* pka_b2    = (const float*)d_in[26];
  const float* acid_w1   = (const float*)d_in[27];
  const float* acid_b1   = (const float*)d_in[28];
  const float* acid_w2   = (const float*)d_in[29];
  const float* acid_b2   = (const float*)d_in[30];
  float* out = (float*)d_out;

  const int N = in_sizes[0] / ADIM;
  const int E = in_sizes[1] / 2;

  // workspace carve
  char* base = (char*)d_ws;
  size_t off = 0;
  auto carve = [&](size_t bytes) -> void* {
    void* p = base + off;
    off = (off + bytes + 255) & ~(size_t)255;
    return p;
  };
  float*     h_f   = (float*)    carve((size_t)N * HD * sizeof(float));
  float*     agg   = (float*)    carve((size_t)N * HD * sizeof(float));
  float*     cntf  = (float*)    carve((size_t)N * sizeof(float));
  _Float16*  h16   = (_Float16*) carve((size_t)N * HD * sizeof(_Float16));
  _Float16*  wT    = (_Float16*) carve((size_t)(16384 * 13 + 32768 * 4) * sizeof(_Float16));

  _Float16* encw2T  = wT;                       // [128][128]
  _Float16* edgew2T = encw2T  + 16384;          // 4 x [128][128]
  _Float16* msgw1T  = edgew2T + 4 * 16384;
  _Float16* msgw2T  = msgw1T  + 4 * 16384;
  _Float16* gatewT  = msgw2T  + 4 * 16384;      // 4 x [128][256]

  // weight conversion (f32 [K][N] -> f16 [N][K])
  to_f16T_kernel<<<64, 256, 0, stream>>>(enc_w2, encw2T, HD, HD);
  for (int i = 0; i < NLAY; ++i){
    to_f16T_kernel<<<64, 256, 0, stream>>>(edge_w2 + i * HD * HD, edgew2T + i * 16384, HD, HD);
    to_f16T_kernel<<<64, 256, 0, stream>>>(msg_w1  + i * HD * HD, msgw1T  + i * 16384, HD, HD);
    to_f16T_kernel<<<64, 256, 0, stream>>>(msg_w2  + i * HD * HD, msgw2T  + i * 16384, HD, HD);
    to_f16T_kernel<<<128, 256, 0, stream>>>(gate_w + i * 2 * HD * HD, gatewT + i * 32768,
                                            2 * HD, HD);
  }

  // in-degree counts
  fill0_kernel<<<(N + 255) / 256, 256, 0, stream>>>(cntf, N);
  count_kernel<<<(E + 255) / 256, 256, 0, stream>>>(ei, cntf, E);

  // encoder
  encoder_kernel<<<(N + 15) / 16, 256, 0, stream>>>(x, enc_w1, enc_b1, encw2T, enc_b2,
                                                    h_f, h16, N);

  // message-passing layers
  for (int i = 0; i < NLAY; ++i){
    fill0_kernel<<<(N * HD + 255) / 256, 256, 0, stream>>>(agg, N * HD);
    edge_layer_kernel<<<(E + 15) / 16, 256, 0, stream>>>(
        ei, edge_attr,
        edge_w1 + i * EDIM * HD, edge_b1 + i * HD, edgew2T + i * 16384, edge_b2 + i * HD,
        msgw1T + i * 16384, msg_b1 + i * HD, msgw2T + i * 16384, msg_b2 + i * HD,
        h16, agg, E);
    node_update_kernel<<<(N + 15) / 16, 256, 0, stream>>>(
        h_f, h16, agg, cntf,
        gatewT + i * 32768, gate_b + i * HD, ln_g + i * HD, ln_b + i * HD, N);
  }

  // output heads
  heads_kernel<<<(N + 63) / 64, 256, 0, stream>>>(
      h_f,
      ion_w1, ion_b1, ion_w2, ion_b2,
      pka_w1, pka_b1, pka_w2, pka_b2,
      acid_w1, acid_b1, acid_w2, acid_b2,
      out, N);
}
